// QMixer_13563506720937
// MI455X (gfx1250) — matrix operand order repaired and checked by reading, not compiled, not run on hardware
//
#include <hip/hip_runtime.h>

#define NB 32768          // BS*T rows
#define N_AGENTS 32
#define STATE_DIM 512
#define EMBED_DIM 64
#define HYPER 256
#define ROWS_PER_WG 128   // 8 waves x 16-row private M-slices

// stage-2 unified B: [w1_0 | wf_0 | hb_w | v0_w], N=640, K=512 -> 40 tiles
#define S2_NTILES 40
// stage-3 unified B: [w1_2 | wf_2], N=2112, K=256 -> 132 tiles
#define S3_NTILES 132

typedef __attribute__((ext_vector_type(16))) _Float16 v16h;
typedef __attribute__((ext_vector_type(8)))  float    v8f;

union Frag { unsigned u[8]; uint4 q[2]; v16h v; };

__device__ __forceinline__ v8f vzero8() {
  v8f z = {0.f, 0.f, 0.f, 0.f, 0.f, 0.f, 0.f, 0.f};
  return z;
}

__device__ __forceinline__ unsigned pk2(float a, float b) {
  union { _Float16 h[2]; unsigned u; } x;
  x.h[0] = (_Float16)a; x.h[1] = (_Float16)b;
  return x.u;
}

// ---------------------------------------------------------------------------
// Prep: swizzle f32 W[N][K] (row-major, used as x @ W^T) into a unified WMMA
// B-fragment stream at tile offset nt0 within a virtual matrix of NtGlobal
// 16-wide N tiles. Fragment order: dword idx = ((kb*NtGlobal+nt)*32+lane)*8+v,
// lane L -> N = nt*16+(L&15), K-half = (L>=16?8:0), VGPR v packs K=base+2v+(v>=4?8:0),+1.
// ---------------------------------------------------------------------------
__global__ void qmix_swizzle(const float* __restrict__ W, unsigned* __restrict__ dst,
                             int N, int K, int NtGlobal, int nt0) {
  int total = (N * K) >> 1;
  int NtLoc = N >> 4;
  for (int idx = blockIdx.x * blockDim.x + threadIdx.x; idx < total;
       idx += gridDim.x * blockDim.x) {
    int v    = idx & 7;
    int lane = (idx >> 3) & 31;
    int rest = idx >> 8;
    int ntl  = rest % NtLoc;
    int kb   = rest / NtLoc;
    int n  = ntl * 16 + (lane & 15);
    int k0 = kb * 32 + ((lane & 16) ? 8 : 0) + 2 * v + ((v & 4) ? 8 : 0);
    size_t di = (((size_t)kb * NtGlobal + nt0 + ntl) * 32 + lane) * 8 + v;
    dst[di] = pk2(W[(size_t)n * K + k0], W[(size_t)n * K + k0 + 1]);
  }
}

// A fragment from LDS f16 row-major (rowDw dwords per row). 16B-aligned.
__device__ __forceinline__ void load_a_frag(Frag* a, const unsigned* __restrict__ Ad,
                                            int rowDw, int m, int half, int kb) {
  const uint4* p = (const uint4*)(Ad + m * rowDw + kb * 16 + half * 4);
  a->q[0] = p[0];
  a->q[1] = p[2];
}

// B fragment from the pre-swizzled global stream (shared by all waves -> WGP$).
__device__ __forceinline__ void load_b_frag(Frag* b, const unsigned* __restrict__ Bsw,
                                            int Ntiles, int kb, int nt, int lane) {
  const uint4* p = (const uint4*)(Bsw) + (size_t)((kb * Ntiles + nt) * 32 + lane) * 2;
  b->q[0] = p[0];
  b->q[1] = p[1];
}

// ---------------------------------------------------------------------------
// Fused QMIX mixer: one WG = 128 rows; wave w privately owns rows [w*16,w*16+16).
// All waves walk the SAME B tile stream -> B served via WGP$. Wave-private
// everything: no atomics, no barriers. 144 KB LDS -> 2 WGs/WGP (4 waves/SIMD).
// b1 / vh / w_final never touch LDS: their C-fragment layout already matches
// the stage-4 consumption layout, so they live in registers (hp / vp / wf).
// ---------------------------------------------------------------------------
__global__ __launch_bounds__(256) void qmix_main(
    const float* __restrict__ agent_qs, const float* __restrict__ states,
    const unsigned* __restrict__ s2sw,  const unsigned* __restrict__ s3sw,
    const float* __restrict__ b1_0v,    const float* __restrict__ b1_2v,
    const float* __restrict__ bf_0v,    const float* __restrict__ bf_2v,
    const float* __restrict__ hb_bv,    const float* __restrict__ v0_bv,
    const float* __restrict__ v2_wv,    const float* __restrict__ v2_bv,
    float* __restrict__ out) {
  __shared__ __align__(16) unsigned char smem[147456];     // 144 KB
  unsigned* H1d = (unsigned*)smem;       // 16384 dw : h1 f16 [128][256]
  unsigned* HFd = H1d + 16384;           // 16384 dw : hf f16 [128][256]
  float* QF = (float*)(HFd + 16384);     //  4096 f : q [128][32]

  const int tid  = threadIdx.x;
  const int lane = tid & 31;
  const int wave = tid >> 5;
  const int b0   = blockIdx.x * ROWS_PER_WG;
  const int mrow = lane & 15;
  const int half = lane >> 4;
  const int nl   = lane & 15;

  // wave-private LDS regions
  unsigned* H1w = H1d + wave * 2048;     // [16][128 dw]
  unsigned* HFw = HFd + wave * 2048;
  float* QFw = QF + wave * 512;          // [16][32]

  // ---- stage 1: q to wave-private LDS; S rows straight into A registers ----
  {
    const float* qp = agent_qs + (size_t)(b0 + wave * 16) * N_AGENTS;
    for (int i = lane; i < 512; i += 32) QFw[i] = qp[i];
  }
  Frag aS[16];                           // full-K A operand (S, f16) in 128 VGPRs
  {
    const float* sp = states + (size_t)(b0 + wave * 16 + mrow) * STATE_DIM + half * 8;
    #pragma unroll
    for (int kb = 0; kb < 16; ++kb) {
      const float4* p = (const float4*)(sp + kb * 32);
      float4 f0 = p[0], f1 = p[1], g0 = p[4], g1 = p[5];
      aS[kb].u[0] = pk2(f0.x, f0.y); aS[kb].u[1] = pk2(f0.z, f0.w);
      aS[kb].u[2] = pk2(f1.x, f1.y); aS[kb].u[3] = pk2(f1.z, f1.w);
      aS[kb].u[4] = pk2(g0.x, g0.y); aS[kb].u[5] = pk2(g0.z, g0.w);
      aS[kb].u[6] = pk2(g1.x, g1.y); aS[kb].u[7] = pk2(g1.z, g1.w);
    }
  }

  v8f hp[4];                             // hidden pre-act: hp[e0][r] @ (m,e)
  float vp[8];                           // per-lane partial of sum(vh * v2_w)
  #pragma unroll
  for (int r = 0; r < 8; ++r) vp[r] = 0.f;

  // ---- stage 2: 10 groups of 4 tiles (h1|hf|b1|vh) for this wave's rows.
  //      kb outer / tile inner: 4 independent WMMA chains share aS[kb].
  #pragma unroll 1
  for (int tg = 0; tg < 10; ++tg) {
    v8f c[4];
    #pragma unroll
    for (int i = 0; i < 4; ++i) c[i] = vzero8();
    #pragma unroll
    for (int kb = 0; kb < 16; ++kb) {
      #pragma unroll
      for (int i = 0; i < 4; ++i) {
        Frag b;
        load_b_frag(&b, s2sw, S2_NTILES, kb, tg * 4 + i, lane);
        c[i] = __builtin_amdgcn_wmma_f32_16x16x32_f16(
            false, aS[kb].v, false, b.v, (short)0, c[i], false, false);
      }
    }
    // epilogues (group never straddles a kind boundary: 16/32/36 are mult of 4)
    #pragma unroll
    for (int i = 0; i < 4; ++i) {
      int t = tg * 4 + i;
      if (t < 16) {                      // h1 = relu -> f16 LDS (A of stage 3)
        float bb = b1_0v[t * 16 + nl];
        _Float16* dst = (_Float16*)H1w;
        #pragma unroll
        for (int r = 0; r < 8; ++r) {
          float x = c[i][r] + bb; x = x > 0.f ? x : 0.f;
          dst[(r + half * 8) * 256 + t * 16 + nl] = (_Float16)x;
        }
      } else if (t < 32) {               // hf = relu -> f16 LDS
        int nt = t - 16;
        float bb = bf_0v[nt * 16 + nl];
        _Float16* dst = (_Float16*)HFw;
        #pragma unroll
        for (int r = 0; r < 8; ++r) {
          float x = c[i][r] + bb; x = x > 0.f ? x : 0.f;
          dst[(r + half * 8) * 256 + nt * 16 + nl] = (_Float16)x;
        }
      } else if (t < 36) {               // b1 -> initializes hidden registers
        int e0 = t - 32;                 // tile == e0-slice: layout matches hp
        float bb = hb_bv[e0 * 16 + nl];
        #pragma unroll
        for (int r = 0; r < 8; ++r)
          hp[e0][r] = c[i][r] + bb;
      } else {                           // vh: fold relu(.)*v2_w into vp
        int e0 = t - 36;
        float bb = v0_bv[e0 * 16 + nl];
        float vw = v2_wv[e0 * 16 + nl];
        #pragma unroll
        for (int r = 0; r < 8; ++r) {
          float x = c[i][r] + bb; x = x > 0.f ? x : 0.f;
          vp[r] += x * vw;
        }
      }
    }
  }

  // ---- stage 3: w1[.,2048] fused with q-contraction (32 groups of 4 tiles),
  //      then |w_final| into registers. A (= h1 / hf) register-cached once.
  v8f wf[4];                             // |w_final|: wf[e0][r] @ (m,e)
  {
    Frag ah[8];
    #pragma unroll
    for (int k8 = 0; k8 < 8; ++k8)
      load_a_frag(&ah[k8], H1w, 128, mrow, half, k8);

    #pragma unroll 1
    for (int jt = 0; jt < 32; ++jt) {    // jt == agent id: tiles jt*4 .. jt*4+3
      v8f c[4];
      #pragma unroll
      for (int i = 0; i < 4; ++i) c[i] = vzero8();
      #pragma unroll
      for (int k8 = 0; k8 < 8; ++k8) {
        #pragma unroll
        for (int q4 = 0; q4 < 4; ++q4) { // e0-slice = q4*16: 4 indep chains
          Frag b;
          load_b_frag(&b, s3sw, S3_NTILES, k8, jt * 4 + q4, lane);
          c[q4] = __builtin_amdgcn_wmma_f32_16x16x32_f16(
              false, ah[k8].v, false, b.v, (short)0, c[q4], false, false);
        }
      }
      float qv[8];
      #pragma unroll
      for (int r = 0; r < 8; ++r)
        qv[r] = QFw[(r + half * 8) * 32 + jt];
      #pragma unroll
      for (int q4 = 0; q4 < 4; ++q4) {
        float bb = b1_2v[(jt * 4 + q4) * 16 + nl];
        #pragma unroll
        for (int r = 0; r < 8; ++r)
          hp[q4][r] += qv[r] * fabsf(c[q4][r] + bb);   // hidden += q * |w1|
      }
    }

    // |w_final| tiles 128..131 as one 4-chain group -> registers only
    Frag af[8];
    #pragma unroll
    for (int k8 = 0; k8 < 8; ++k8)
      load_a_frag(&af[k8], HFw, 128, mrow, half, k8);
    {
      v8f c[4];
      #pragma unroll
      for (int i = 0; i < 4; ++i) c[i] = vzero8();
      #pragma unroll
      for (int k8 = 0; k8 < 8; ++k8) {
        #pragma unroll
        for (int w4 = 0; w4 < 4; ++w4) {
          Frag b;
          load_b_frag(&b, s3sw, S3_NTILES, k8, 128 + w4, lane);
          c[w4] = __builtin_amdgcn_wmma_f32_16x16x32_f16(
              false, af[k8].v, false, b.v, (short)0, c[w4], false, false);
        }
      }
      #pragma unroll
      for (int w4 = 0; w4 < 4; ++w4) {
        float bb = bf_2v[w4 * 16 + nl];
        #pragma unroll
        for (int r = 0; r < 8; ++r)
          wf[w4][r] = fabsf(c[w4][r] + bb);
      }
    }
  }

  // ---- stage 4: elu + dot reductions entirely in registers + shfl ----
  {
    float vb = v2_bv[0];
    float p[8];
    #pragma unroll
    for (int r = 0; r < 8; ++r) {
      float sum = vp[r];
      #pragma unroll
      for (int e0 = 0; e0 < 4; ++e0) {
        float h = hp[e0][r];
        h = h > 0.f ? h : (expf(h) - 1.f);               // elu
        sum += h * wf[e0][r];
      }
      // reduce over the 16 lanes of this half (bit4 of lane untouched)
      sum += __shfl_xor(sum, 1);
      sum += __shfl_xor(sum, 2);
      sum += __shfl_xor(sum, 4);
      sum += __shfl_xor(sum, 8);
      p[r] = sum;
    }
    if (nl == 0) {
      #pragma unroll
      for (int r = 0; r < 8; ++r)
        out[b0 + wave * 16 + half * 8 + r] = p[r] + vb;
    }
  }
}

extern "C" void kernel_launch(void* const* d_in, const int* in_sizes, int n_in,
                              void* d_out, int out_size, void* d_ws, size_t ws_size,
                              hipStream_t stream) {
  (void)in_sizes; (void)n_in; (void)out_size; (void)ws_size;
  const float* agent_qs = (const float*)d_in[0];
  const float* states   = (const float*)d_in[1];
  const float* w1_0 = (const float*)d_in[2];
  const float* b1_0 = (const float*)d_in[3];
  const float* w1_2 = (const float*)d_in[4];
  const float* b1_2 = (const float*)d_in[5];
  const float* wf_0 = (const float*)d_in[6];
  const float* bf_0 = (const float*)d_in[7];
  const float* wf_2 = (const float*)d_in[8];
  const float* bf_2 = (const float*)d_in[9];
  const float* hb_w = (const float*)d_in[10];
  const float* hb_b = (const float*)d_in[11];
  const float* v0_w = (const float*)d_in[12];
  const float* v0_b = (const float*)d_in[13];
  const float* v2_w = (const float*)d_in[14];
  const float* v2_b = (const float*)d_in[15];

  // workspace: two unified f16-swizzled B streams (dword counts)
  unsigned* ws   = (unsigned*)d_ws;
  unsigned* s2sw = ws;               // 640*512/2  = 163840 dw
  unsigned* s3sw = ws + 163840;      // 2112*256/2 = 270336 dw  (total ~1.7 MB)

  auto launch_sw = [&](const float* W, unsigned* dst, int N, int K,
                       int NtGlobal, int nt0) {
    int total  = (N * K) >> 1;
    int blocks = (total + 255) >> 8;
    if (blocks > 1024) blocks = 1024;
    qmix_swizzle<<<blocks, 256, 0, stream>>>(W, dst, N, K, NtGlobal, nt0);
  };
  // stage-2 stream: K=512, 40 tiles
  launch_sw(w1_0, s2sw, 256, 512, S2_NTILES, 0);
  launch_sw(wf_0, s2sw, 256, 512, S2_NTILES, 16);
  launch_sw(hb_w, s2sw, 64,  512, S2_NTILES, 32);
  launch_sw(v0_w, s2sw, 64,  512, S2_NTILES, 36);
  // stage-3 stream: K=256, 132 tiles
  launch_sw(w1_2, s3sw, 2048, 256, S3_NTILES, 0);
  launch_sw(wf_2, s3sw, 64,   256, S3_NTILES, 128);

  qmix_main<<<NB / ROWS_PER_WG, 256, 0, stream>>>(agent_qs, states, s2sw, s3sw,
      b1_0, b1_2, bf_0, bf_2, hb_b, v0_b, v2_w, v2_b, (float*)d_out);
}
